// QLoRALinear_30691836297881
// MI455X (gfx1250) — matrix operand order repaired and checked by reading, not compiled, not run on hardware
//
#include <hip/hip_runtime.h>

// ---------------------------------------------------------------------------
// QLoRA 2-bit dequant GEMM for gfx1250 (MI455X), wave32 + v_wmma_f32_16x16x32_f16
// Packed-f16 dequant: 2 codes per v_pk_fma_f16.
// ---------------------------------------------------------------------------

typedef __attribute__((ext_vector_type(16))) _Float16 v16h;
typedef __attribute__((ext_vector_type(8)))  _Float16 v8h;
typedef __attribute__((ext_vector_type(4)))  _Float16 v4h;
typedef __attribute__((ext_vector_type(2)))  _Float16 v2h;
typedef __attribute__((ext_vector_type(8)))  float    v8f;
typedef __attribute__((ext_vector_type(4)))  float    v4f;
typedef __attribute__((ext_vector_type(4)))  int      v4i;
typedef __attribute__((ext_vector_type(4)))  unsigned v4u;

#define OUT_F   8192
#define IN_F    8192
#define TOKENS  256
#define RANK    16
// per out-row: IN_F/4 = 2048 packed bytes (each stored as one int32)
#define ROW_PACK 2048
// LDS tile row stride in halves (32 data + 8 pad = 80 bytes, keeps 16B alignment)
#define LSTR 40
#define KSTEPS (IN_F / 32)   // 256

// ---------------------------------------------------------------------------
// Kernel 1: x (f32) -> f16 staging buffer
// ---------------------------------------------------------------------------
__global__ __launch_bounds__(256) void qlora_convert_x(
    const float* __restrict__ x, _Float16* __restrict__ x16)
{
    size_t i = ((size_t)blockIdx.x * 256 + threadIdx.x) * 4;
    v4f v = *(const v4f*)(x + i);
    v4h h;
    h[0] = (_Float16)v[0];
    h[1] = (_Float16)v[1];
    h[2] = (_Float16)v[2];
    h[3] = (_Float16)v[3];
    *(v4h*)(x16 + i) = h;
}

// ---------------------------------------------------------------------------
// Kernel 2: t[256,16] = x @ lora_a^T   (one block per token)
// ---------------------------------------------------------------------------
__global__ __launch_bounds__(256) void qlora_lora_t(
    const float* __restrict__ x, const float* __restrict__ lora_a,
    float* __restrict__ t)
{
    __shared__ float red[256];
    const int m   = blockIdx.x;
    const int r   = threadIdx.x & 15;   // rank index
    const int seg = threadIdx.x >> 4;   // k segment 0..15 (512 elems each)

    const float* xr = x      + (size_t)m * IN_F + seg * 512;
    const float* ar = lora_a + (size_t)r * IN_F + seg * 512;
    float s = 0.f;
    #pragma unroll 4
    for (int k = 0; k < 512; k += 4) {
        v4f xv = *(const v4f*)(xr + k);
        v4f av = *(const v4f*)(ar + k);
        s += xv[0]*av[0] + xv[1]*av[1] + xv[2]*av[2] + xv[3]*av[3];
    }
    red[threadIdx.x] = s;
    __syncthreads();
    if (threadIdx.x < 16) {
        float acc = 0.f;
        #pragma unroll
        for (int sgi = 0; sgi < 16; ++sgi) acc += red[sgi * 16 + threadIdx.x];
        t[m * RANK + threadIdx.x] = acc;
    }
}

// ---------------------------------------------------------------------------
// Kernel 3: main dequant-GEMM. Block tile: M=128 x N=128, K-step 32.
// 8 waves in 2(M) x 4(N) grid; each wave: 4x2 fragments of 16x16.
// ---------------------------------------------------------------------------
struct Fetch {
    v8h a0, a1;     // 16 halves of x16 for this thread's A-tile slot
    v4i p;          // 4 packed bytes (int32 each) of weight_q2
    float norm;     // group norm (exact f16 value promoted to f32)
};

// Dequantize two 2-bit codes packed as f16 {1024+q0, 1024+q1} in `u`:
//   q = d - 1024 (exact, Sterbenz), val = q*(2/3*norm) - norm
// -> one v_pk_add_f16 + one v_pk_fma_f16.
__device__ inline unsigned dq2(unsigned u, v2h s2, v2h n2, v2h c1024) {
    v2h d = __builtin_bit_cast(v2h, u);
    d = d - c1024;
    v2h r = d * s2 - n2;
    return __builtin_bit_cast(unsigned, r);
}

__global__ __launch_bounds__(256) void qlora_main(
    const _Float16* __restrict__ x16,
    const int*      __restrict__ wq,
    const _Float16* __restrict__ wnorm,
    const float*    __restrict__ bias,
    const float*    __restrict__ lora_b,
    const float*    __restrict__ t,
    float*          __restrict__ out)
{
    __shared__ __align__(16) _Float16 sA[2][128 * LSTR];
    __shared__ __align__(16) _Float16 sB[2][128 * LSTR];

    const int tid  = threadIdx.x;
    const int lane = tid & 31;
    const int wid  = tid >> 5;

    const int nTile = blockIdx.x * 128;   // N tile base (64 tiles)
    const int mTile = blockIdx.y * 128;   // M tile base (2 tiles)

    // loader role: each thread owns half a tile row (16 halves / 4 packed ints)
    const int lrow  = tid >> 1;           // 0..127
    const int lhalf = tid & 1;            // 0..1
    const int nG = nTile + lrow;          // global out-feature row for B loads
    const int mG = mTile + lrow;          // global token row for A loads

    // fragment addressing (documented wave32 WMMA layouts)
    const int aM  = lane & 15;            // A: row within 16
    const int aKb = (lane >> 4) * 8;      // A: K sub-base {0,8}
    const int bN  = lane & 15;            // B: column within 16
    const int bK0 = (lane >> 4) * 8;      // B: K sub-base {0,8}
    const int mW  = (wid >> 2) * 64;      // wave M offset in tile
    const int nW  = (wid & 3) * 32;       // wave N offset in tile

    const v2h c1024 = {(_Float16)1024.0f, (_Float16)1024.0f};

    v8f acc[4][2];
    #pragma unroll
    for (int i = 0; i < 4; ++i)
        #pragma unroll
        for (int j = 0; j < 2; ++j) acc[i][j] = {};

    auto global_fetch = [&](int kk) {
        Fetch f;
        const _Float16* xsrc = x16 + (size_t)mG * IN_F + kk * 32 + lhalf * 16;
        f.a0 = *(const v8h*)(xsrc);
        f.a1 = *(const v8h*)(xsrc + 8);
        f.p  = *(const v4i*)(wq + (size_t)nG * ROW_PACK + kk * 8 + lhalf * 4);
        f.norm = (float)wnorm[(size_t)nG * (IN_F / 128) + (kk >> 2)];
        return f;
    };

    auto stage_store = [&](const Fetch& f, int buf) {
        // A tile: row-major [m][k], 32 halves per row
        _Float16* da = &sA[buf][lrow * LSTR + lhalf * 16];
        *(v8h*)(da)     = f.a0;
        *(v8h*)(da + 8) = f.a1;

        // B tile: dequantize 4 packed bytes -> 16 f16 values (k-contiguous),
        // two codes per packed-f16 op.
        const _Float16 sh = (_Float16)(f.norm * (2.0f / 3.0f));
        const _Float16 nh = (_Float16)f.norm;
        const v2h s2 = {sh, sh};
        const v2h n2 = {nh, nh};

        v4u w0, w1;
        #pragma unroll
        for (int b = 0; b < 4; ++b) {
            unsigned bb = (unsigned)f.p[b];
            // {1024+q0, 1024+q1} and {1024+q2, 1024+q3} as packed f16 bits
            unsigned u0 = 0x64006400u | (bb & 3u)          | ((bb & 0x0Cu) << 14);
            unsigned u1 = 0x64006400u | ((bb >> 4) & 3u)   | ((bb & 0xC0u) << 10);
            unsigned r0 = dq2(u0, s2, n2, c1024);
            unsigned r1 = dq2(u1, s2, n2, c1024);
            if (b < 2) { w0[(b & 1) * 2] = r0; w0[(b & 1) * 2 + 1] = r1; }
            else       { w1[(b & 1) * 2] = r0; w1[(b & 1) * 2 + 1] = r1; }
        }
        v4u* db = (v4u*)&sB[buf][lrow * LSTR + lhalf * 16];
        db[0] = w0;
        db[1] = w1;
    };

    auto compute_stage = [&](int buf) {
        v16h a[4], b[2];
        #pragma unroll
        for (int i = 0; i < 4; ++i) {
            const _Float16* base = &sA[buf][(mW + i * 16 + aM) * LSTR];
            v8h lo = *(const v8h*)(base + aKb);
            v8h hi = *(const v8h*)(base + aKb + 16);
            a[i] = __builtin_shufflevector(lo, hi, 0,1,2,3,4,5,6,7,8,9,10,11,12,13,14,15);
        }
        #pragma unroll
        for (int j = 0; j < 2; ++j) {
            const _Float16* base = &sB[buf][(nW + j * 16 + bN) * LSTR];
            v8h lo = *(const v8h*)(base + bK0);
            v8h hi = *(const v8h*)(base + bK0 + 16);
            b[j] = __builtin_shufflevector(lo, hi, 0,1,2,3,4,5,6,7,8,9,10,11,12,13,14,15);
        }
        #pragma unroll
        for (int i = 0; i < 4; ++i)
            #pragma unroll
            for (int j = 0; j < 2; ++j)
                acc[i][j] = __builtin_amdgcn_wmma_f32_16x16x32_f16(
                    false, a[i], false, b[j], (short)0, acc[i][j], false, false);
    };

    // ---- pipelined K loop (double-buffered LDS, one barrier per step) ----
    {
        Fetch f0 = global_fetch(0);
        stage_store(f0, 0);
    }
    __syncthreads();

    for (int kk = 0; kk < KSTEPS; ++kk) {
        const int cur = kk & 1;
        const bool has = (kk + 1) < KSTEPS;
        Fetch nf;
        if (has) {
            nf = global_fetch(kk + 1);
            if (kk + 2 < KSTEPS)   // warm L2/WGP$ two steps ahead on the weight stream
                __builtin_prefetch(wq + (size_t)nG * ROW_PACK + (kk + 2) * 8 + lhalf * 4, 0, 1);
        }
        compute_stage(cur);
        if (has) stage_store(nf, cur ^ 1);
        __syncthreads();
    }

    // ---- epilogue: + bias + rank-16 LoRA, using C/D layout (lane->N, vgpr->M)
    const int rl = lane >> 4;   // 0/1 -> M offset 0/8
    const int cl = lane & 15;   // N within 16
    #pragma unroll
    for (int i = 0; i < 4; ++i) {
        #pragma unroll
        for (int j = 0; j < 2; ++j) {
            const int n = nTile + nW + j * 16 + cl;
            const float bv = bias[n];
            const v4f* lbp = (const v4f*)(lora_b + (size_t)n * RANK);
            v4f l0 = lbp[0], l1 = lbp[1], l2 = lbp[2], l3 = lbp[3];
            #pragma unroll
            for (int r = 0; r < 8; ++r) {
                const int m = mTile + mW + i * 16 + rl * 8 + r;
                const v4f* tp = (const v4f*)(t + (size_t)m * RANK);
                v4f t0 = tp[0], t1 = tp[1], t2 = tp[2], t3 = tp[3];
                float s = bv + acc[i][j][r];
                s += t0[0]*l0[0] + t0[1]*l0[1] + t0[2]*l0[2] + t0[3]*l0[3];
                s += t1[0]*l1[0] + t1[1]*l1[1] + t1[2]*l1[2] + t1[3]*l1[3];
                s += t2[0]*l2[0] + t2[1]*l2[1] + t2[2]*l2[2] + t2[3]*l2[3];
                s += t3[0]*l3[0] + t3[1]*l3[1] + t3[2]*l3[2] + t3[3]*l3[3];
                out[(size_t)m * OUT_F + n] = s;
            }
        }
    }
}

// ---------------------------------------------------------------------------
// Launcher
// ---------------------------------------------------------------------------
extern "C" void kernel_launch(void* const* d_in, const int* in_sizes, int n_in,
                              void* d_out, int out_size, void* d_ws, size_t ws_size,
                              hipStream_t stream) {
    const float*    x    = (const float*)d_in[0];
    const int*      wq   = (const int*)d_in[1];
    const _Float16* wn   = (const _Float16*)d_in[2];
    const float*    bias = (const float*)d_in[3];
    const float*    la   = (const float*)d_in[4];
    const float*    lb   = (const float*)d_in[5];
    float*          out  = (float*)d_out;

    _Float16* x16 = (_Float16*)d_ws;                                   // 4 MiB
    float*    t   = (float*)((char*)d_ws + (size_t)TOKENS * IN_F * 2); // 16 KiB

    qlora_convert_x<<<(TOKENS * IN_F) / (256 * 4), 256, 0, stream>>>(x, x16);
    qlora_lora_t<<<TOKENS, 256, 0, stream>>>(x, la, t);
    qlora_main<<<dim3(OUT_F / 128, TOKENS / 128), 256, 0, stream>>>(
        x16, wq, wn, bias, lb, t, out);
}
